// ASpTLinear_33483565039897
// MI455X (gfx1250) — matrix operand order repaired and checked by reading, not compiled, not run on hardware
//
#include <hip/hip_runtime.h>
#include <hip/hip_bf16.h>

// out[M,N] = x[M,K] * W^T + bias ; M=8192, N=4096, K=4096, fp32 in/out.
// Pipeline (when ws_size allows):
//   1) convert W, x fp32 -> bf16 once into d_ws (memory-bound prepass)
//   2) bf16 WMMA GEMM feeding LDS via global_load_async_to_lds_b128 (ASYNCcnt
//      DMA), double-buffered so DMA of tile t+1 overlaps WMMA on tile t.
// Fallback: single-kernel convert-in-loop version (round-2) if ws too small.

typedef __attribute__((ext_vector_type(16))) __bf16 v16bf;
typedef __attribute__((ext_vector_type(8)))  __bf16 v8bf;
typedef __attribute__((ext_vector_type(4)))  __bf16 v4bf;
typedef __attribute__((ext_vector_type(8)))  float  v8f;

#define M_TOT 8192
#define N_TOT 4096
#define K_TOT 4096
#define BM 128
#define BN 128
#define BK 64
#define LDSK 72   // padded bf16 row stride: 144 bytes (16B-aligned, bank-spreading)

#define WB_BYTES (N_TOT * K_TOT * 2)               // 32 MB bf16 weight
#define XB_BYTES (M_TOT * K_TOT * 2)               // 64 MB bf16 activations
#define WS_NEED  ((size_t)WB_BYTES + (size_t)XB_BYTES)

__device__ __forceinline__ v4bf cvt4_bf16(float4 f) {
    v4bf r;
    r[0] = (__bf16)f.x;  r[1] = (__bf16)f.y;
    r[2] = (__bf16)f.z;  r[3] = (__bf16)f.w;
    return r;
}

// ---------------------------------------------------------------- prepass ---
__global__ __launch_bounds__(256)
void cvt_f32_to_bf16_kernel(const float4* __restrict__ in,
                            v4bf* __restrict__ out, int n4) {
    int i = blockIdx.x * 256 + threadIdx.x;
    if (i < n4) out[i] = cvt4_bf16(in[i]);
}

// ------------------------------------------------- async-DMA helpers --------
__device__ __forceinline__ void async_load_b128(unsigned lds_addr, unsigned voff,
                                                const void* base) {
    // GVS mode: mem = SADDR(64) + VADDR(32) ; LDS dest per lane from vdst VGPR.
    asm volatile("global_load_async_to_lds_b128 %0, %1, %2 offset:0"
                 :: "v"(lds_addr), "v"(voff), "s"(base) : "memory");
}
__device__ __forceinline__ void wait_async_le8() {
    asm volatile("s_wait_asynccnt 0x8" ::: "memory");
}
__device__ __forceinline__ void wait_async_0() {
    asm volatile("s_wait_asynccnt 0x0" ::: "memory");
}

// -------------------------------------------------- main bf16 GEMM (async) --
__global__ __launch_bounds__(256)
void gemm_bf16_async_wmma_kernel(const __bf16* __restrict__ xb,
                                 const __bf16* __restrict__ wb,
                                 const float* __restrict__ bias,
                                 float* __restrict__ out)
{
    __shared__ __bf16 lA[2][BM * LDSK];
    __shared__ __bf16 lB[2][BN * LDSK];

    const int tid  = threadIdx.x;
    const int lane = tid & 31;
    const int wave = tid >> 5;
    const int wm   = wave >> 2;   // 0..1 : 64-row slab
    const int wn   = wave & 3;    // 0..3 : 32-col slab
    const int l15  = lane & 15;
    const int hi   = lane >> 4;

    const int brow = blockIdx.y * BM;
    const int bcol = blockIdx.x * BN;

    // DMA chunk map: tile = 128 rows x 64 bf16 = 128 x 8 x 16B chunks.
    // 256 threads x 4 chunks cover one matrix tile.
    unsigned goffA[4], goffB[4];                 // global byte offsets (bf16)
    unsigned ldsA[2][4], ldsB[2][4];             // per-lane LDS byte dests
#pragma unroll
    for (int p = 0; p < 4; ++p) {
        int lin = p * 256 + tid;
        int row = lin >> 3;                      // 0..127
        int cq  = lin & 7;                       // 16-byte chunk in row
        goffA[p] = (unsigned)(((brow + row) * K_TOT + cq * 8) * 2);
        goffB[p] = (unsigned)(((bcol + row) * K_TOT + cq * 8) * 2);
#pragma unroll
        for (int b = 0; b < 2; ++b) {
            ldsA[b][p] = (unsigned)(size_t)&lA[b][row * LDSK + cq * 8];
            ldsB[b][p] = (unsigned)(size_t)&lB[b][row * LDSK + cq * 8];
        }
    }

    v8f zero = {0.f, 0.f, 0.f, 0.f, 0.f, 0.f, 0.f, 0.f};
    v8f acc[4][2];
#pragma unroll
    for (int m = 0; m < 4; ++m)
#pragma unroll
        for (int n = 0; n < 2; ++n)
            acc[m][n] = zero;

    constexpr int TILES = K_TOT / BK;            // 64

    // Prologue: DMA tile 0 into buffer 0.
#pragma unroll
    for (int p = 0; p < 4; ++p) {
        async_load_b128(ldsA[0][p], goffA[p], xb);
        async_load_b128(ldsB[0][p], goffB[p], wb);
    }
#pragma unroll
    for (int p = 0; p < 4; ++p) { goffA[p] += BK * 2; goffB[p] += BK * 2; }

    for (int t = 0; t < TILES; ++t) {
        const int cur = t & 1;
        if (t + 1 < TILES) {
            const int nxt = cur ^ 1;
#pragma unroll
            for (int p = 0; p < 4; ++p) {
                async_load_b128(ldsA[nxt][p], goffA[p], xb);
                async_load_b128(ldsB[nxt][p], goffB[p], wb);
            }
#pragma unroll
            for (int p = 0; p < 4; ++p) { goffA[p] += BK * 2; goffB[p] += BK * 2; }
            wait_async_le8();   // in-order completion => tile t landed
        } else {
            wait_async_0();
        }
        __syncthreads();        // all waves' DMAs for tile t visible

        const __bf16* curA = lA[cur];
        const __bf16* curB = lB[cur];

#pragma unroll
        for (int s = 0; s < 2; ++s) {            // two K=32 slices
            v16bf afrag[4], bfrag[2];
#pragma unroll
            for (int m = 0; m < 4; ++m) {
                int row = wm * 64 + m * 16 + l15;
                const v8bf* p0 = (const v8bf*)(curA + row * LDSK + s * 32 + hi * 8);
                const v8bf* p1 = (const v8bf*)(curA + row * LDSK + s * 32 + hi * 8 + 16);
                afrag[m] = __builtin_shufflevector(p0[0], p1[0],
                    0, 1, 2, 3, 4, 5, 6, 7, 8, 9, 10, 11, 12, 13, 14, 15);
            }
#pragma unroll
            for (int n = 0; n < 2; ++n) {
                int col = wn * 32 + n * 16 + l15;
                const v8bf* p0 = (const v8bf*)(curB + col * LDSK + s * 32 + hi * 8);
                const v8bf* p1 = (const v8bf*)(curB + col * LDSK + s * 32 + hi * 8 + 16);
                bfrag[n] = __builtin_shufflevector(p0[0], p1[0],
                    0, 1, 2, 3, 4, 5, 6, 7, 8, 9, 10, 11, 12, 13, 14, 15);
            }
#pragma unroll
            for (int m = 0; m < 4; ++m)
#pragma unroll
                for (int n = 0; n < 2; ++n)
                    acc[m][n] = __builtin_amdgcn_wmma_f32_16x16x32_bf16(
                        false, afrag[m], false, bfrag[n],
                        (short)0, acc[m][n], false, false);
        }
        __syncthreads();        // done reading buf before it is DMA-overwritten
    }

    float bv[2];
#pragma unroll
    for (int n = 0; n < 2; ++n)
        bv[n] = bias[bcol + wn * 32 + n * 16 + l15];

#pragma unroll
    for (int m = 0; m < 4; ++m)
#pragma unroll
        for (int n = 0; n < 2; ++n) {
            int col = bcol + wn * 32 + n * 16 + l15;
#pragma unroll
            for (int v = 0; v < 8; ++v) {
                int row = brow + wm * 64 + m * 16 + hi * 8 + v;
                out[(size_t)row * N_TOT + col] = acc[m][n][v] + bv[n];
            }
        }
}

// ----------------------------------------- fallback: convert-in-loop GEMM ---
__global__ __launch_bounds__(256)
void gemm_f32in_wmma_kernel(const float* __restrict__ x,
                            const float* __restrict__ w,
                            const float* __restrict__ bias,
                            float* __restrict__ out)
{
    __shared__ __bf16 lA[BM * LDSK];
    __shared__ __bf16 lB[BN * LDSK];

    const int tid  = threadIdx.x;
    const int lane = tid & 31;
    const int wave = tid >> 5;
    const int wm   = wave >> 2;
    const int wn   = wave & 3;
    const int l15  = lane & 15;
    const int hi   = lane >> 4;

    const int brow = blockIdx.y * BM;
    const int bcol = blockIdx.x * BN;

    unsigned offA[8], offB[8];
    int rowS[8], kqS[8];
#pragma unroll
    for (int p = 0; p < 8; ++p) {
        int lin = p * 256 + tid;
        rowS[p] = lin >> 4;
        kqS[p]  = lin & 15;
        offA[p] = (unsigned)(((brow + rowS[p]) * K_TOT + kqS[p] * 4) * 4);
        offB[p] = (unsigned)(((bcol + rowS[p]) * K_TOT + kqS[p] * 4) * 4);
    }

    v8f zero = {0.f, 0.f, 0.f, 0.f, 0.f, 0.f, 0.f, 0.f};
    v8f acc[4][2];
#pragma unroll
    for (int m = 0; m < 4; ++m)
#pragma unroll
        for (int n = 0; n < 2; ++n)
            acc[m][n] = zero;

    const char* xB = (const char*)x;
    const char* wB = (const char*)w;

    for (int kit = 0; kit < K_TOT / BK; ++kit) {
        float4 ra[8], rb[8];
#pragma unroll
        for (int p = 0; p < 8; ++p) {
            ra[p] = *(const float4*)(xB + offA[p]);
            rb[p] = *(const float4*)(wB + offB[p]);
            offA[p] += BK * 4; offB[p] += BK * 4;
        }
        __syncthreads();
#pragma unroll
        for (int p = 0; p < 8; ++p) {
            *(v4bf*)(lA + rowS[p] * LDSK + kqS[p] * 4) = cvt4_bf16(ra[p]);
            *(v4bf*)(lB + rowS[p] * LDSK + kqS[p] * 4) = cvt4_bf16(rb[p]);
        }
        __syncthreads();
#pragma unroll
        for (int s = 0; s < 2; ++s) {
            v16bf afrag[4], bfrag[2];
#pragma unroll
            for (int m = 0; m < 4; ++m) {
                int row = wm * 64 + m * 16 + l15;
                const v8bf* p0 = (const v8bf*)(lA + row * LDSK + s * 32 + hi * 8);
                const v8bf* p1 = (const v8bf*)(lA + row * LDSK + s * 32 + hi * 8 + 16);
                afrag[m] = __builtin_shufflevector(p0[0], p1[0],
                    0, 1, 2, 3, 4, 5, 6, 7, 8, 9, 10, 11, 12, 13, 14, 15);
            }
#pragma unroll
            for (int n = 0; n < 2; ++n) {
                int col = wn * 32 + n * 16 + l15;
                const v8bf* p0 = (const v8bf*)(lB + col * LDSK + s * 32 + hi * 8);
                const v8bf* p1 = (const v8bf*)(lB + col * LDSK + s * 32 + hi * 8 + 16);
                bfrag[n] = __builtin_shufflevector(p0[0], p1[0],
                    0, 1, 2, 3, 4, 5, 6, 7, 8, 9, 10, 11, 12, 13, 14, 15);
            }
#pragma unroll
            for (int m = 0; m < 4; ++m)
#pragma unroll
                for (int n = 0; n < 2; ++n)
                    acc[m][n] = __builtin_amdgcn_wmma_f32_16x16x32_bf16(
                        false, afrag[m], false, bfrag[n],
                        (short)0, acc[m][n], false, false);
        }
    }

    float bv[2];
#pragma unroll
    for (int n = 0; n < 2; ++n)
        bv[n] = bias[bcol + wn * 32 + n * 16 + l15];
#pragma unroll
    for (int m = 0; m < 4; ++m)
#pragma unroll
        for (int n = 0; n < 2; ++n) {
            int col = bcol + wn * 32 + n * 16 + l15;
#pragma unroll
            for (int v = 0; v < 8; ++v) {
                int row = brow + wm * 64 + m * 16 + hi * 8 + v;
                out[(size_t)row * N_TOT + col] = acc[m][n][v] + bv[n];
            }
        }
}

extern "C" void kernel_launch(void* const* d_in, const int* in_sizes, int n_in,
                              void* d_out, int out_size, void* d_ws, size_t ws_size,
                              hipStream_t stream) {
    const float* x    = (const float*)d_in[0];  // [4,2048,4096]
    const float* wgt  = (const float*)d_in[1];  // [4096,4096]
    const float* bias = (const float*)d_in[2];  // [4096]
    float* out = (float*)d_out;                 // [4,2048,4096]

    dim3 grid(N_TOT / BN, M_TOT / BM);          // (32, 64)

    if (ws_size >= WS_NEED) {
        __bf16* wb = (__bf16*)d_ws;
        __bf16* xb = (__bf16*)((char*)d_ws + WB_BYTES);

        int n4w = N_TOT * K_TOT / 4;            // 4,194,304 float4
        int n4x = M_TOT * K_TOT / 4;            // 8,388,608 float4
        cvt_f32_to_bf16_kernel<<<n4w / 256, 256, 0, stream>>>(
            (const float4*)wgt, (v4bf*)wb, n4w);
        cvt_f32_to_bf16_kernel<<<n4x / 256, 256, 0, stream>>>(
            (const float4*)x, (v4bf*)xb, n4x);

        gemm_bf16_async_wmma_kernel<<<grid, dim3(256), 0, stream>>>(xb, wb, bias, out);
    } else {
        gemm_f32in_wmma_kernel<<<grid, dim3(256), 0, stream>>>(x, wgt, bias, out);
    }
}
